// StructInformedAttention_41437844471886
// MI455X (gfx1250) — matrix operand order repaired and checked by reading, not compiled, not run on hardware
//
#include <hip/hip_runtime.h>
#include <hip/hip_bf16.h>

// ---------------------------------------------------------------------------
// Types for CDNA5 WMMA (wave32): v16bf = A/B operand (16x32 / 32x16 bf16),
// v8f = 16x16 f32 C/D accumulator.
// ---------------------------------------------------------------------------
typedef __bf16 v16bf __attribute__((ext_vector_type(16)));
typedef __bf16 v8bf  __attribute__((ext_vector_type(8)));
typedef float  v8f   __attribute__((ext_vector_type(8)));

union bfu { v16bf v; v8bf h[2]; };

static __device__ __forceinline__ v8f vzero8() {
    v8f z = {0.f, 0.f, 0.f, 0.f, 0.f, 0.f, 0.f, 0.f};
    return z;
}

static __device__ __forceinline__ v8f wmma_bf16(v16bf a, v16bf b, v8f c) {
    // D = A(16x32 bf16) * B(32x16 bf16) + C(f32)
    return __builtin_amdgcn_wmma_f32_16x16x32_bf16(
        /*neg_a=*/false, a, /*neg_b=*/false, b,
        /*c_mod=*/(short)0, c, /*reuse_a=*/false, /*reuse_b=*/false);
}

// fp32 -> bf16 round-to-nearest-even
static __device__ __forceinline__ unsigned short f2bf(float f) {
    unsigned int u = __float_as_uint(f);
    unsigned int r = (u + 0x7FFFu + ((u >> 16) & 1u)) >> 16;
    return (unsigned short)r;
}

// Async 16B copy global -> LDS (CDNA5 GLOBAL_LOAD_ASYNC_TO_LDS_B128,
// tracked with ASYNCcnt). LDS address = addr[31:0] of the generic pointer
// (ISA 10.2: LDS aperture truncates to low 32 bits).
static __device__ __forceinline__ void async_copy_b128(const void* gptr, void* lptr) {
    unsigned lds_off = (unsigned)(unsigned long long)(uintptr_t)lptr;
    asm volatile("global_load_async_to_lds_b128 %0, %1, off"
                 :: "v"(lds_off), "v"(gptr) : "memory");
}

// ---------------------------------------------------------------------------
// Problem constants
// ---------------------------------------------------------------------------
#define PB   2
#define PS   2048
#define PHID 1024
#define PNH  16
#define PHD  64
#define PM   (PB * PS)        // 4096 GEMM rows

#define KSTRIDE 72            // padded LDS row stride (halves) for K tiles
#define VSTRIDE 136           // padded LDS row stride (halves) for V tiles
#define PSTRIDE 136           // padded LDS row stride (halves) for P scratch

// ---------------------------------------------------------------------------
// Kernel 1: fp32 -> bf16 cast
// ---------------------------------------------------------------------------
__global__ void __launch_bounds__(256)
cast_f32_to_bf16(const float* __restrict__ in, unsigned short* __restrict__ out, int n) {
    int i = blockIdx.x * 256 + threadIdx.x;
    if (i < n) out[i] = f2bf(in[i]);
}

// ---------------------------------------------------------------------------
// Kernel 2: out = X[4096,1024] @ W[1024,1024]^T + bias, bf16 in/out, f32 acc.
// Block = 4 waves; block tile 64(M) x 64(N); each wave: 16(M) x 64(N).
// transposed==0: store [B,NH,S,HD]; transposed==1: store [B,NH,HD,S].
// ---------------------------------------------------------------------------
__global__ void __launch_bounds__(128)
qkv_gemm_kernel(const unsigned short* __restrict__ X,
                const unsigned short* __restrict__ W,
                const float* __restrict__ bias,
                unsigned short* __restrict__ outp,
                int transposed) {
    const int lane = threadIdx.x & 31;
    const int wave = threadIdx.x >> 5;
    const int mrow = lane & 15;   // A: M index, B: N index, C/D: N index
    const int half = lane >> 4;   // lane half selects K/M sub-blocks
    const int m0 = blockIdx.y * 64 + wave * 16;
    const int n0 = blockIdx.x * 64;

    v8f acc[4];
#pragma unroll
    for (int t = 0; t < 4; ++t) acc[t] = vzero8();

    // A row base for this lane (contiguous K order used here: lane half offsets K by 16)
    const unsigned short* arow = X + (size_t)(m0 + mrow) * PHID + half * 16;

    for (int k0 = 0; k0 < PHID; k0 += 32) {
        bfu a;
        a.h[0] = *(const v8bf*)(const void*)(arow + k0);        // K = k0 + half*16 + 0..7
        a.h[1] = *(const v8bf*)(const void*)(arow + k0 + 8);    // K = k0 + half*16 + 8..15
#pragma unroll
        for (int t = 0; t < 4; ++t) {
            // B operand: n = n0+t*16+mrow fixed per lane, K contiguous (B[k][n]=W[n][k])
            const unsigned short* bp =
                W + (size_t)(n0 + t * 16 + mrow) * PHID + k0 + half * 16;
            v16bf bf = *(const v16bf*)(const void*)bp;
            acc[t] = wmma_bf16(a.v, bf, acc[t]);
        }
    }

    // Bias + store in head-separated layout.
#pragma unroll
    for (int t = 0; t < 4; ++t) {
        const int n = n0 + t * 16 + mrow;
        const float bv = bias[n];
        const int hh = n >> 6;     // head
        const int d  = n & 63;     // dim-in-head
#pragma unroll
        for (int r = 0; r < 8; ++r) {
            const int m  = m0 + half * 8 + r;   // global row (b*S + s)
            const int bb = m >> 11;
            const int s  = m & 2047;
            const float val = acc[t][r] + bv;
            size_t idx;
            if (transposed)
                idx = (((size_t)(bb * PNH + hh) * PHD + d) * PS + s);
            else
                idx = (((size_t)(bb * PNH + hh) * PS + s) * PHD + d);
            outp[idx] = f2bf(val);
        }
    }
}

// ---------------------------------------------------------------------------
// Stage one 128-key chunk of K (128 x 64 bf16) and V^T (64 x 128 bf16) into
// LDS via async copies. 128 threads x 16 ops = 2048 x 16B = 32 KB.
// ---------------------------------------------------------------------------
static __device__ __forceinline__ void stage_chunk(
    const unsigned short* __restrict__ Kh,   // [S, HD]
    const unsigned short* __restrict__ Vh,   // [HD, S]
    int kb, unsigned short* kbuf, unsigned short* vbuf, int tid) {
#pragma unroll
    for (int i = 0; i < 8; ++i) {            // K: 128 rows x 8 segs of 16B... (8 segs/row)
        const int c = i * 128 + tid;         // 0..1023
        const int row = c >> 3;              // key row 0..127
        const int seg = c & 7;               // 16B segment within 128B row
        async_copy_b128(Kh + (size_t)(kb + row) * PHD + seg * 8,
                        kbuf + row * KSTRIDE + seg * 8);
    }
#pragma unroll
    for (int i = 0; i < 8; ++i) {            // V^T: 64 rows x 16 segs of 16B
        const int c = i * 128 + tid;         // 0..1023
        const int row = c >> 4;              // d row 0..63
        const int seg = c & 15;              // 16B segment within 256B row
        async_copy_b128(Vh + (size_t)row * PS + kb + seg * 8,
                        vbuf + row * VSTRIDE + seg * 8);
    }
}

// ---------------------------------------------------------------------------
// Kernel 3: flash attention.
// grid = (S/64, NH, B); block = 4 waves; each wave owns 16 queries.
// K/V chunks (128 keys) are double-buffered in LDS via async-to-LDS copies
// shared by all 4 waves. Per chunk: QK^T (16 WMMA), masks + online softmax,
// P->LDS (bf16), P@V (16 WMMA) accumulating into fp32 D-fragments.
// ---------------------------------------------------------------------------
__global__ void __launch_bounds__(128)
attn_kernel(const unsigned short* __restrict__ Qb,   // [B,NH,S,HD] bf16
            const unsigned short* __restrict__ Kb,   // [B,NH,S,HD] bf16
            const unsigned short* __restrict__ Vtb,  // [B,NH,HD,S] bf16
            const float* __restrict__ amask,         // [B,S]
            const float* __restrict__ smask,         // [B,S,S]
            const float* __restrict__ gate,          // [NH]
            float* __restrict__ out)                 // [B,S,HID]
{
    __shared__ __align__(16) unsigned short Kl[2][128 * KSTRIDE];  // 2 x 18 KB
    __shared__ __align__(16) unsigned short Vl[2][64 * VSTRIDE];   // 2 x 17 KB
    __shared__ __align__(16) unsigned short pshm[4 * 16 * PSTRIDE]; // 17 KB

    const int tid  = threadIdx.x;
    const int lane = tid & 31;
    const int wave = tid >> 5;
    const int mrow = lane & 15;
    const int half = lane >> 4;
    const int b = blockIdx.z;
    const int h = blockIdx.y;
    const int q0 = blockIdx.x * 64 + wave * 16;
    const float g = gate[h];

    const size_t headoff = (size_t)(b * PNH + h) * PS * PHD;
    const unsigned short* Qh = Qb + headoff;
    const unsigned short* Kh = Kb + headoff;
    const unsigned short* Vh = Vtb + headoff;
    unsigned short* pw = pshm + wave * 16 * PSTRIDE;

    // Q fragments (A layout), K-dim = 64 -> two 16x32 fragments
    bfu qf[2];
    const unsigned short* qrow = Qh + (size_t)(q0 + mrow) * PHD + half * 16;
#pragma unroll
    for (int kk = 0; kk < 2; ++kk) {
        qf[kk].h[0] = *(const v8bf*)(const void*)(qrow + kk * 32);
        qf[kk].h[1] = *(const v8bf*)(const void*)(qrow + kk * 32 + 8);
    }

    float mi[8], li[8];
#pragma unroll
    for (int r = 0; r < 8; ++r) { mi[r] = -3.0e38f; li[r] = 0.f; }
    v8f oacc[4];
#pragma unroll
    for (int t = 0; t < 4; ++t) oacc[t] = vzero8();

    const float* smbase = smask + (size_t)b * PS * PS;
    const float* ambase = amask + (size_t)b * PS;

    // Prime the pipeline: chunk 0 -> buffer 0.
    stage_chunk(Kh, Vh, 0, Kl[0], Vl[0], tid);

    for (int ci = 0; ci < PS / 128; ++ci) {
        const int kb = ci * 128;
        const int bi = ci & 1;

        // Prefetch next chunk into the other buffer (overlaps with compute
        // of this chunk); then wait for THIS chunk's 16 async ops (same-
        // counter async loads complete in order -> <=16 leaves only the
        // just-issued prefetch outstanding).
        if (ci + 1 < PS / 128) {
            stage_chunk(Kh, Vh, kb + 128, Kl[bi ^ 1], Vl[bi ^ 1], tid);
            asm volatile("s_wait_asynccnt 0x10" ::: "memory");
        } else {
            asm volatile("s_wait_asynccnt 0x0" ::: "memory");
        }
        __syncthreads();   // current buffers visible to all 4 waves

        // ---- scores: 8 key sub-tiles of 16 (K fragments from LDS) ----
        v8f sc[8];
#pragma unroll
        for (int j = 0; j < 8; ++j) {
            const unsigned short* krow =
                &Kl[bi][(j * 16 + mrow) * KSTRIDE + half * 16];
            bfu kf0, kf1;
            kf0.h[0] = *(const v8bf*)(const void*)(krow);
            kf0.h[1] = *(const v8bf*)(const void*)(krow + 8);
            kf1.h[0] = *(const v8bf*)(const void*)(krow + 32);
            kf1.h[1] = *(const v8bf*)(const void*)(krow + 40);
            v8f a = vzero8();
            a = wmma_bf16(qf[0].v, kf0.v, a);
            a = wmma_bf16(qf[1].v, kf1.v, a);
            sc[j] = a;
        }
        // ---- scale + masks ----
#pragma unroll
        for (int j = 0; j < 8; ++j) {
            const int key = kb + j * 16 + mrow;
            const float am = ambase[key];
#pragma unroll
            for (int r = 0; r < 8; ++r) {
                const int q = q0 + half * 8 + r;
                sc[j][r] = sc[j][r] * 0.125f + am + smbase[(size_t)q * PS + key] * g;
            }
        }
        // ---- row max (across 8 sub-tiles, then across 16 lanes of the half) ----
        float scalef[8];
#pragma unroll
        for (int r = 0; r < 8; ++r) {
            float v = sc[0][r];
#pragma unroll
            for (int j = 1; j < 8; ++j) v = fmaxf(v, sc[j][r]);
            v = fmaxf(v, __shfl_xor(v, 1, 16));
            v = fmaxf(v, __shfl_xor(v, 2, 16));
            v = fmaxf(v, __shfl_xor(v, 4, 16));
            v = fmaxf(v, __shfl_xor(v, 8, 16));
            const float mn = fmaxf(mi[r], v);
            scalef[r] = __expf(mi[r] - mn);
            mi[r] = mn;
        }
        // ---- exponentiate, stash P (bf16) in LDS, row sums ----
        float rs[8];
#pragma unroll
        for (int r = 0; r < 8; ++r) rs[r] = 0.f;
#pragma unroll
        for (int j = 0; j < 8; ++j) {
#pragma unroll
            for (int r = 0; r < 8; ++r) {
                const float p = __expf(sc[j][r] - mi[r]);
                rs[r] += p;
                pw[(half * 8 + r) * PSTRIDE + j * 16 + mrow] = f2bf(p);
            }
        }
#pragma unroll
        for (int r = 0; r < 8; ++r) {
            float v = rs[r];
            v += __shfl_xor(v, 1, 16);
            v += __shfl_xor(v, 2, 16);
            v += __shfl_xor(v, 4, 16);
            v += __shfl_xor(v, 8, 16);
            li[r] = li[r] * scalef[r] + v;
#pragma unroll
            for (int t = 0; t < 4; ++t) oacc[t][r] *= scalef[r];
        }
        // cross-lane LDS transpose: make sure this wave's P stores landed
        asm volatile("s_wait_dscnt 0" ::: "memory");
        // ---- P @ V : 4 d-tiles x 4 K-steps of 32 keys (V from LDS) ----
#pragma unroll
        for (int t = 0; t < 4; ++t) {
#pragma unroll
            for (int kk = 0; kk < 4; ++kk) {
                bfu pf, vf;
                const unsigned short* pl = pw + mrow * PSTRIDE + kk * 32 + half * 16;
                pf.h[0] = *(const v8bf*)(const void*)(pl);
                pf.h[1] = *(const v8bf*)(const void*)(pl + 8);
                const unsigned short* vrow =
                    &Vl[bi][(t * 16 + mrow) * VSTRIDE + kk * 32 + half * 16];
                vf.h[0] = *(const v8bf*)(const void*)(vrow);
                vf.h[1] = *(const v8bf*)(const void*)(vrow + 8);
                oacc[t] = wmma_bf16(pf.v, vf.v, oacc[t]);
            }
        }
        __syncthreads();   // all waves done reading buffers before reuse
    }

    // ---- normalize and store ctx -> [B,S,HID] ----
#pragma unroll
    for (int t = 0; t < 4; ++t) {
        const int col = h * PHD + t * 16 + mrow;
#pragma unroll
        for (int r = 0; r < 8; ++r) {
            const int q = q0 + half * 8 + r;
            out[((size_t)b * PS + q) * PHID + col] = oacc[t][r] / li[r];
        }
    }
}

// ---------------------------------------------------------------------------
// Host launcher
// ---------------------------------------------------------------------------
extern "C" void kernel_launch(void* const* d_in, const int* in_sizes, int n_in,
                              void* d_out, int out_size, void* d_ws, size_t ws_size,
                              hipStream_t stream) {
    (void)in_sizes; (void)n_in; (void)out_size; (void)ws_size;
    const float* hidden = (const float*)d_in[0];
    const float* amask  = (const float*)d_in[1];
    const float* smask  = (const float*)d_in[2];
    const float* Wq     = (const float*)d_in[3];
    const float* bq     = (const float*)d_in[4];
    const float* Wk     = (const float*)d_in[5];
    const float* bk     = (const float*)d_in[6];
    const float* Wv     = (const float*)d_in[7];
    const float* bv     = (const float*)d_in[8];
    const float* gate   = (const float*)d_in[9];
    float* out = (float*)d_out;

    // Workspace layout (bf16 halves), ~40 MB total
    unsigned short* ws  = (unsigned short*)d_ws;
    unsigned short* hb  = ws;                                  // 4096*1024
    unsigned short* wqb = hb  + (size_t)PM * PHID;             // 1024*1024
    unsigned short* wkb = wqb + (size_t)PHID * PHID;
    unsigned short* wvb = wkb + (size_t)PHID * PHID;
    unsigned short* Qb  = wvb + (size_t)PHID * PHID;           // B*NH*S*HD
    unsigned short* Kb  = Qb  + (size_t)PB * PNH * PS * PHD;
    unsigned short* Vtb = Kb  + (size_t)PB * PNH * PS * PHD;

    const int nh = PM * PHID;     // 4,194,304
    const int nw = PHID * PHID;   // 1,048,576
    cast_f32_to_bf16<<<nh / 256, 256, 0, stream>>>(hidden, hb, nh);
    cast_f32_to_bf16<<<nw / 256, 256, 0, stream>>>(Wq, wqb, nw);
    cast_f32_to_bf16<<<nw / 256, 256, 0, stream>>>(Wk, wkb, nw);
    cast_f32_to_bf16<<<nw / 256, 256, 0, stream>>>(Wv, wvb, nw);

    dim3 gg(PHID / 64, PM / 64);  // (16, 64)
    qkv_gemm_kernel<<<gg, 128, 0, stream>>>(hb, wqb, bq, Qb, 0);
    qkv_gemm_kernel<<<gg, 128, 0, stream>>>(hb, wkb, bk, Kb, 0);
    qkv_gemm_kernel<<<gg, 128, 0, stream>>>(hb, wvb, bv, Vtb, 1);

    dim3 ga(PS / 64, PNH, PB);    // (32, 16, 2)
    attn_kernel<<<ga, 128, 0, stream>>>(Qb, Kb, Vtb, amask, smask, gate, out);
}
